// TransformerEncoder_9689446220356
// MI455X (gfx1250) — matrix operand order repaired and checked by reading, not compiled, not run on hardware
//
#include <hip/hip_runtime.h>
#include <hip/hip_bf16.h>

typedef _Float16 half_t;
typedef __attribute__((ext_vector_type(16))) _Float16 v16h;
typedef __attribute__((ext_vector_type(8)))  float    v8f;
typedef __attribute__((ext_vector_type(4)))  unsigned int u32x4;

union F16x16 { v16h h; u32x4 u[2]; };
union H8 { u32x4 u; half_t h[8]; };

#define WMMA_F16(A, B, C) \
    __builtin_amdgcn_wmma_f32_16x16x32_f16(false, (A), false, (B), (short)0, (C), false, false)

// --------- CDNA5 async global->LDS copy (GLOBAL_LOAD_ASYNC_TO_LDS_B128) ------
__device__ __forceinline__ void async_copy_b128(const void* gaddr, void* lds) {
    unsigned l = (unsigned)(uintptr_t)lds;            // low 32b of generic = LDS offset
    unsigned long long g = (unsigned long long)(uintptr_t)gaddr;
    asm volatile("global_load_async_to_lds_b128 %0, %1, off"
                 :: "v"(l), "v"(g) : "memory");
}
__device__ __forceinline__ void wait_async0() {
    asm volatile("s_wait_asynccnt 0x0" ::: "memory");
}

// ---------------------------------------------------------------- f32 -> f16
__global__ __launch_bounds__(256) void k_f2h(const float* __restrict__ s,
                                             half_t* __restrict__ d, int n) {
    int i = blockIdx.x * 256 + threadIdx.x;
    if (i < n) d[i] = (half_t)s[i];
}

// --------------------------------------------- combined dist + path bias (B,N,N)
__global__ __launch_bounds__(256) void k_bias(const float* __restrict__ dist,
                                              const int* __restrict__ paths,
                                              const float* __restrict__ ef,
                                              const float* __restrict__ pp,
                                              const float* __restrict__ dparams,
                                              float* __restrict__ biasOut) {
    __shared__ __align__(16) float sPP[8][16];
    int t = threadIdx.x;
    if (t < 128) sPP[t >> 4][t & 15] = pp[t];
    __syncthreads();
    int i = blockIdx.x * 256 + t;            // < 8*256*256
    float acc = 0.f;
    const int* pth = paths + (size_t)i * 8;
#pragma unroll
    for (int p = 0; p < 8; p++) {
        int id = pth[p];
        if (id >= 0) {
            const float4* e = (const float4*)(ef + (size_t)id * 16);
            const float4* w = (const float4*)sPP[p];
#pragma unroll
            for (int q = 0; q < 4; q++) {
                float4 a = e[q], b = w[q];
                acc += a.x * b.x + a.y * b.y + a.z * b.z + a.w * b.w;
            }
        }
    }
    biasOut[i] = dist[i] * dparams[0] + dparams[1] + acc * 0.125f;
}

// --------------------------------------------------- xp = x + deg*W + b (f16)
__global__ __launch_bounds__(256) void k_prep(const float* __restrict__ x,
                                              const float* __restrict__ degrees,
                                              const float* __restrict__ deg_W,
                                              const float* __restrict__ deg_b,
                                              half_t* __restrict__ xp) {
    int i = blockIdx.x * 256 + threadIdx.x;  // < 2048*128
    int row = i >> 7, d = i & 127;
    xp[i] = (half_t)(x[i] + degrees[row] * deg_W[d] + deg_b[d]);
}

// ----------------------------------------------------------- layernorm -> f16
__global__ __launch_bounds__(256) void k_ln(const float* __restrict__ h,
                                            const float* __restrict__ g,
                                            const float* __restrict__ b,
                                            half_t* __restrict__ z) {
    __shared__ float red[256];
    int row = blockIdx.x, t = threadIdx.x;
    float v = h[(size_t)row * 256 + t];
    red[t] = v; __syncthreads();
    for (int s = 128; s > 0; s >>= 1) { if (t < s) red[t] += red[t + s]; __syncthreads(); }
    float mean = red[0] * (1.f / 256.f);
    __syncthreads();
    float d = v - mean;
    red[t] = d * d; __syncthreads();
    for (int s = 128; s > 0; s >>= 1) { if (t < s) red[t] += red[t + s]; __syncthreads(); }
    float var = red[0] * (1.f / 256.f);
    z[(size_t)row * 256 + t] = (half_t)(d * rsqrtf(var + 1e-5f) * g[t] + b[t]);
}

// ------------------------------------------- pipelined WMMA GEMM, 64x128 tile
// A tile staged via GLOBAL_LOAD_ASYNC_TO_LDS_B128 (ASYNCcnt); W tile transposed
// via registers. Double-buffered LDS; async(i+1) issued after barrier(i) so it
// overlaps the WMMAs of tile i while targeting the vacated buffer.
__global__ __launch_bounds__(256) void k_gemm(const half_t* __restrict__ A,
                                              const half_t* __restrict__ W,
                                              const float* __restrict__ bias,
                                              const float* __restrict__ res,
                                              float* __restrict__ outF,
                                              half_t* __restrict__ outH,
                                              int M, int K, int Nc, int relu) {
    __shared__ __align__(16) half_t sA[2][64][40];    // A tiles, row-major, padded
    __shared__ __align__(16) half_t sW[2][128][40];   // W tiles, transposed [n][k]
    int t = threadIdx.x, lane = t & 31, wid = t >> 5;
    int mblk = blockIdx.y * 64, nblk = blockIdx.x * 128;
    int wm = (wid >> 1) * 16, wn = (wid & 1) * 64;
    int m = lane & 15, hi = lane >> 4;
    v8f acc[4] = {};

    // staging roles
    int r  = t >> 2, c  = t & 3;     // A: 64 rows x 4 chunks of 8 halves
    int kr = t >> 4, cg = t & 15;    // W: k-rows {kr, kr+16}, 16 col-groups of 8
    const half_t* Ap  = A + (size_t)(mblk + r) * K + c * 8;
    const half_t* Wp0 = W + (size_t)kr * Nc + nblk + cg * 8;
    const half_t* Wp1 = Wp0 + (size_t)16 * Nc;
    const size_t wstep = (size_t)32 * Nc;

    __builtin_prefetch(bias + nblk + wn + m, 0, 0);
    if (res) __builtin_prefetch(res + (size_t)(mblk + wm) * Nc + nblk + wn + m, 0, 0);

    int nk = K >> 5;
    async_copy_b128(Ap, &sA[0][r][c * 8]);            // tile 0 A -> LDS (async)
    u32x4 gw0 = *(const u32x4*)(Wp0);
    u32x4 gw1 = *(const u32x4*)(Wp1);

    for (int i = 0; i < nk; i++) {
        int buf = i & 1;
        {   // publish W tile i (transposed)
            H8 w0, w1; w0.u = gw0; w1.u = gw1;
#pragma unroll
            for (int q = 0; q < 8; q++) {
                sW[buf][cg * 8 + q][kr]      = w0.h[q];
                sW[buf][cg * 8 + q][kr + 16] = w1.h[q];
            }
        }
        wait_async0();      // A tile i landed in sA[buf]
        __syncthreads();    // tiles visible; all waves vacated the other buffer
        if (i + 1 < nk) {   // overlap next A copy + W prefetch with WMMAs below
            async_copy_b128(Ap + (size_t)(i + 1) * 32, &sA[buf ^ 1][r][c * 8]);
            gw0 = *(const u32x4*)(Wp0 + (size_t)(i + 1) * wstep);
            gw1 = *(const u32x4*)(Wp1 + (size_t)(i + 1) * wstep);
        }
        F16x16 af, bf[4];
        af.u[0] = *(const u32x4*)(&sA[buf][wm + m][8 * hi]);
        af.u[1] = *(const u32x4*)(&sA[buf][wm + m][16 + 8 * hi]);
#pragma unroll
        for (int a = 0; a < 4; a++) {
            const half_t* bp = &sW[buf][wn + a * 16 + m][8 * hi];
            bf[a].u[0] = *(const u32x4*)(bp);
            bf[a].u[1] = *(const u32x4*)(bp + 16);
        }
#pragma unroll
        for (int a = 0; a < 4; a++)
            acc[a] = WMMA_F16(af.h, bf[a].h, acc[a]);
    }
#pragma unroll
    for (int a = 0; a < 4; a++) {
#pragma unroll
        for (int j = 0; j < 8; j++) {
            int grow = mblk + wm + j + 8 * hi;
            int gc = nblk + wn + a * 16 + m;
            size_t idx = (size_t)grow * Nc + gc;
            float v = acc[a][j] + bias[gc];
            if (res)  v += res[idx];
            if (relu) v = fmaxf(v, 0.f);
            if (outF) outF[idx] = v;
            if (outH) outH[idx] = (half_t)v;
        }
    }
}

// ---------------------------------------------------- fused attention per (b,h)
__global__ __launch_bounds__(512) void k_attn(const half_t* __restrict__ qkv,
                                              const float* __restrict__ bias,
                                              half_t* __restrict__ vals) {
    __shared__ __align__(16) half_t sK[256][32];    // [key][dim]
    __shared__ __align__(16) half_t sVT[32][264];   // [dim][key], padded
    __shared__ __align__(16) half_t sP[16][16][32]; // per-wave prob staging
    int t = threadIdx.x, lane = t & 31, wid = t >> 5;
    int bb = blockIdx.x >> 3, hh = blockIdx.x & 7;
    {   // stage K via async copy, V^T via register transpose
        int key = t >> 1, part = t & 1;
        const half_t* base = qkv + ((size_t)(bb * 256 + key) * 768 + hh * 96);
        async_copy_b128(base + 32 + part * 16,     &sK[key][part * 16]);
        async_copy_b128(base + 32 + part * 16 + 8, &sK[key][part * 16 + 8]);
        const half_t* vb = base + 64 + part * 16;
        H8 v0, v1; v0.u = *(const u32x4*)(vb); v1.u = *(const u32x4*)(vb + 8);
#pragma unroll
        for (int q = 0; q < 8; q++) {
            sVT[part * 16 + q][key]     = v0.h[q];
            sVT[part * 16 + 8 + q][key] = v1.h[q];
        }
    }
    wait_async0();
    __syncthreads();
    int m = lane & 15, hi = lane >> 4;
    int qbase = wid * 16;
    F16x16 qf;
    {
        const half_t* qb = qkv + ((size_t)(bb * 256 + qbase + m) * 768 + hh * 96);
        qf.u[0] = *(const u32x4*)(qb + 8 * hi);
        qf.u[1] = *(const u32x4*)(qb + 16 + 8 * hi);
    }
    v8f sc[16];
#pragma unroll
    for (int tt = 0; tt < 16; tt++) {
        F16x16 bf;
        bf.u[0] = *(const u32x4*)(&sK[tt * 16 + m][8 * hi]);
        bf.u[1] = *(const u32x4*)(&sK[tt * 16 + m][8 * hi + 16]);
        v8f zc = {};
        sc[tt] = WMMA_F16(qf.h, bf.h, zc);
    }
    const float scale = 0.17677669529663687f;  // 1/sqrt(32)
    float rmax[8], rsum[8];
#pragma unroll
    for (int j = 0; j < 8; j++) rmax[j] = -1e30f;
#pragma unroll
    for (int tt = 0; tt < 16; tt++)
#pragma unroll
        for (int j = 0; j < 8; j++) {
            float v = sc[tt][j] * scale;
            sc[tt][j] = v;
            rmax[j] = fmaxf(rmax[j], v);
        }
#pragma unroll
    for (int j = 0; j < 8; j++) {
        for (int s = 1; s < 16; s <<= 1) rmax[j] = fmaxf(rmax[j], __shfl_xor(rmax[j], s, 32));
        rsum[j] = 0.f;
    }
#pragma unroll
    for (int tt = 0; tt < 16; tt++)
#pragma unroll
        for (int j = 0; j < 8; j++) {
            float e = __expf(sc[tt][j] - rmax[j]);
            sc[tt][j] = e;
            rsum[j] += e;
        }
#pragma unroll
    for (int j = 0; j < 8; j++) {
        for (int s = 1; s < 16; s <<= 1) rsum[j] += __shfl_xor(rsum[j], s, 32);
        rsum[j] = 1.f / rsum[j];
    }
    v8f oa0 = {}, oa1 = {};
    for (int jp = 0; jp < 8; jp++) {
#pragma unroll
        for (int tt2 = 0; tt2 < 2; tt2++) {
            int tt = jp * 2 + tt2;
#pragma unroll
            for (int j = 0; j < 8; j++) {
                int row = j + 8 * hi;
                int key = tt * 16 + m;
                float pv = sc[tt][j] * rsum[j]
                         + bias[((size_t)(bb * 256 + qbase + row)) * 256 + key];
                sP[wid][row][tt2 * 16 + m] = (half_t)pv;   // post-softmax bias per reference
            }
        }
        F16x16 af, bv0, bv1;                                // DS ops in-order per wave
        af.u[0]  = *(const u32x4*)(&sP[wid][m][8 * hi]);
        af.u[1]  = *(const u32x4*)(&sP[wid][m][16 + 8 * hi]);
        bv0.u[0] = *(const u32x4*)(&sVT[m][jp * 32 + 8 * hi]);
        bv0.u[1] = *(const u32x4*)(&sVT[m][jp * 32 + 8 * hi + 16]);
        bv1.u[0] = *(const u32x4*)(&sVT[m + 16][jp * 32 + 8 * hi]);
        bv1.u[1] = *(const u32x4*)(&sVT[m + 16][jp * 32 + 8 * hi + 16]);
        oa0 = WMMA_F16(af.h, bv0.h, oa0);
        oa1 = WMMA_F16(af.h, bv1.h, oa1);
    }
#pragma unroll
    for (int j = 0; j < 8; j++) {
        int row = qbase + j + 8 * hi;
        size_t base = ((size_t)(bb * 256 + row)) * 256 + hh * 32;
        vals[base + m]      = (half_t)oa0[j];
        vals[base + 16 + m] = (half_t)oa1[j];
    }
}

// ----------------------------------------------------- out = mean_n(h) @ out_W
__global__ __launch_bounds__(256) void k_out(const float* __restrict__ h,
                                             const float* __restrict__ out_W,
                                             const float* __restrict__ out_b,
                                             float* __restrict__ out) {
    __shared__ float red[256];
    int b = blockIdx.x, t = threadIdx.x;
    float acc = 0.f;
    for (int n = 0; n < 256; n++) acc += h[((size_t)(b * 256 + n)) * 256 + t];
    red[t] = acc * out_W[t];
    __syncthreads();
    for (int s = 128; s > 0; s >>= 1) { if (t < s) red[t] += red[t + s]; __syncthreads(); }
    if (t == 0) out[b] = red[0] * (1.f / 256.f) + out_b[0];
}

// ============================================================== orchestration
extern "C" void kernel_launch(void* const* d_in, const int* in_sizes, int n_in,
                              void* d_out, int out_size, void* d_ws, size_t ws_size,
                              hipStream_t stream) {
    const float* x          = (const float*)d_in[0];
    const float* degrees    = (const float*)d_in[1];
    const float* dist_mat   = (const float*)d_in[2];
    const int*   paths      = (const int*)d_in[3];
    const float* edge_feats = (const float*)d_in[4];
    const float* deg_W      = (const float*)d_in[5];
    const float* deg_b      = (const float*)d_in[6];
    const float* emb_W      = (const float*)d_in[7];
    const float* emb_b      = (const float*)d_in[8];
    const float* dist_par   = (const float*)d_in[9];
    const float* path_par   = (const float*)d_in[10];
    const float* qkv_W      = (const float*)d_in[11];
    const float* qkv_b      = (const float*)d_in[12];
    const float* o_W        = (const float*)d_in[13];
    const float* o_b        = (const float*)d_in[14];
    const float* ln1_g      = (const float*)d_in[15];
    const float* ln1_b      = (const float*)d_in[16];
    const float* ln2_g      = (const float*)d_in[17];
    const float* ln2_b      = (const float*)d_in[18];
    const float* ff1_W      = (const float*)d_in[19];
    const float* ff1_b      = (const float*)d_in[20];
    const float* ff2_W      = (const float*)d_in[21];
    const float* ff2_b      = (const float*)d_in[22];
    const float* out_W      = (const float*)d_in[23];
    const float* out_b      = (const float*)d_in[24];
    float* out = (float*)d_out;

    size_t off = 0;
    auto alloc = [&](size_t bytes) -> void* {
        void* r = (char*)d_ws + off;
        off += (bytes + 255) & ~(size_t)255;
        return r;
    };
    const int M = 2048;                       // B*N tokens
    float*  biasbuf = (float*)alloc((size_t)8 * 256 * 256 * 4);
    half_t* xp      = (half_t*)alloc((size_t)M * 128 * 2);
    float*  h       = (float*) alloc((size_t)M * 256 * 4);
    half_t* z       = (half_t*)alloc((size_t)M * 256 * 2);
    half_t* qkvb    = (half_t*)alloc((size_t)M * 768 * 2);
    half_t* valsb   = (half_t*)alloc((size_t)M * 256 * 2);
    half_t* ffb     = (half_t*)alloc((size_t)M * 1024 * 2);
    half_t* wemb    = (half_t*)alloc((size_t)128 * 256 * 2);
    half_t* wqkv    = (half_t*)alloc((size_t)4 * 256 * 768 * 2);
    half_t* wo      = (half_t*)alloc((size_t)4 * 256 * 256 * 2);
    half_t* wff1    = (half_t*)alloc((size_t)4 * 256 * 1024 * 2);
    half_t* wff2    = (half_t*)alloc((size_t)4 * 1024 * 256 * 2);

    auto cvt = [&](const float* s, half_t* d, int n) {
        k_f2h<<<(n + 255) / 256, 256, 0, stream>>>(s, d, n);
    };
    cvt(emb_W, wemb, 128 * 256);
    cvt(qkv_W, wqkv, 4 * 256 * 768);
    cvt(o_W,   wo,   4 * 256 * 256);
    cvt(ff1_W, wff1, 4 * 256 * 1024);
    cvt(ff2_W, wff2, 4 * 1024 * 256);

    k_bias<<<(8 * 256 * 256) / 256, 256, 0, stream>>>(dist_mat, paths, edge_feats,
                                                      path_par, dist_par, biasbuf);
    k_prep<<<(M * 128) / 256, 256, 0, stream>>>(x, degrees, deg_W, deg_b, xp);

    // embedding: h = xp @ emb_W + emb_b
    k_gemm<<<dim3(256 / 128, M / 64), 256, 0, stream>>>(xp, wemb, emb_b, nullptr,
                                                        h, nullptr, M, 128, 256, 0);
    for (int l = 0; l < 4; l++) {
        k_ln<<<M, 256, 0, stream>>>(h, ln1_g + l * 256, ln1_b + l * 256, z);
        k_gemm<<<dim3(768 / 128, M / 64), 256, 0, stream>>>(z, wqkv + (size_t)l * 256 * 768,
                                                            qkv_b + l * 768, nullptr,
                                                            nullptr, qkvb, M, 256, 768, 0);
        k_attn<<<64, 512, 0, stream>>>(qkvb, biasbuf, valsb);
        k_gemm<<<dim3(256 / 128, M / 64), 256, 0, stream>>>(valsb, wo + (size_t)l * 256 * 256,
                                                            o_b + l * 256, h,
                                                            h, nullptr, M, 256, 256, 0);
        k_ln<<<M, 256, 0, stream>>>(h, ln2_g + l * 256, ln2_b + l * 256, z);
        k_gemm<<<dim3(1024 / 128, M / 64), 256, 0, stream>>>(z, wff1 + (size_t)l * 256 * 1024,
                                                             ff1_b + l * 1024, nullptr,
                                                             nullptr, ffb, M, 256, 1024, 1);
        k_gemm<<<dim3(256 / 128, M / 64), 256, 0, stream>>>(ffb, wff2 + (size_t)l * 1024 * 256,
                                                            ff2_b + l * 256, h,
                                                            h, nullptr, M, 1024, 256, 0);
    }
    k_out<<<8, 256, 0, stream>>>(h, out_W, out_b, out);
}
